// SPDRMLR_71751723647146
// MI455X (gfx1250) — matrix operand order repaired and checked by reading, not compiled, not run on hardware
//
#include <hip/hip_runtime.h>
#include <hip/hip_bf16.h>
#include <math.h>

// ---------------------------------------------------------------------------
// SPD Log-Euclidean RMLR for MI455X (gfx1250, wave32, WMMA + async-to-LDS).
//   out[n,c] = <logm(X_n), A_sym_c> - <logm(P_c), A_sym_c>,  <V,W>=tr(VW)
// Pipeline:
//   1) prep_asym   : A_sym (f32 for bias dot, f16 for WMMA B operand)
//   2) jacobi_logm : batched 64x64 symmetric eigendecomp (parallel cyclic
//                    Jacobi, one 64-thread workgroup per matrix, LDS-resident,
//                    padded stride 65 -> conflict-free rows AND columns),
//                    then logX = V diag(log w) V^T, emitted as f16 (X) / f32 (P)
//   3) dot_bias    : b[c] = tr(logP_c A_sym_c)
//   4) rmlr_gemm   : (CHUNK x 4096) x (4096 x 128) GEMM on
//                    v_wmma_f32_16x16x32_f16; B tile staged in LDS via
//                    double-buffered global_load_async_to_lds_b128 (ASYNCcnt),
//                    fused bias subtract.
// X is processed in 4096-matrix chunks so the f16 logX staging tensor (32MB)
// stays hot in the 192MB L2 between producer and consumer kernels.
// ---------------------------------------------------------------------------

typedef __attribute__((ext_vector_type(16))) _Float16 v16h;
typedef __attribute__((ext_vector_type(8)))  _Float16 v8h;
typedef __attribute__((ext_vector_type(8)))  float    v8f;

union H16 { v16h v; v8h h[2]; };

#define NDIM   64
#define NDIM2  4096
#define NMAT   32768
#define NCLS   128
#define LROW   65      // LDS row stride in floats: 65*4B -> bank = i for columns
#define NSWEEP 8
#define CHUNK  4096

// B-tile staging in rmlr_gemm: 128 rows x 32 halves, row stride padded to
// 40 halves (80B) -> 16B-aligned ds_load_b128 and conflict-free column reads.
#define BROW   40
#define BBUF   (NCLS * BROW)   // halves per buffer

// ---------------------------------------------------------------------------
// 1) A_sym[c][i][j] = A[c][max(i,j)][min(i,j)]  (strict-lower + transpose + diag)
// ---------------------------------------------------------------------------
__global__ __launch_bounds__(256) void prep_asym(const float* __restrict__ A,
                                                 float* __restrict__ As32,
                                                 _Float16* __restrict__ As16) {
  const int c = blockIdx.x;
  const float* Ac = A + (size_t)c * NDIM2;
  for (int e = threadIdx.x; e < NDIM2; e += 256) {
    const int i = e >> 6, j = e & 63;
    const int mx = i > j ? i : j, mn = i > j ? j : i;
    const float v = Ac[mx * NDIM + mn];
    As32[(size_t)c * NDIM2 + e] = v;
    As16[(size_t)c * NDIM2 + e] = (_Float16)v;
  }
}

// ---------------------------------------------------------------------------
// 2) Batched SPD logm via parallel cyclic Jacobi.
//    64 threads (2 wave32s) per matrix. Round-robin tournament pairing gives
//    32 disjoint (p,q) rotations per round; row phase then column phase, each
//    staged through registers with workgroup barriers (no RAW races).
// ---------------------------------------------------------------------------
template <bool HALF_OUT>
__global__ __launch_bounds__(64) void jacobi_logm(const float* __restrict__ src,
                                                  void* __restrict__ dst) {
  __shared__ float As[NDIM * LROW];
  __shared__ float Vs[NDIM * LROW];
  __shared__ float lw[NDIM];

  const int t = threadIdx.x;            // 0..63, owns row t / column t
  const size_t mat = blockIdx.x;

  const float* S = src + mat * NDIM2 + (size_t)t * NDIM;
#pragma unroll
  for (int j = 0; j < NDIM; ++j) As[t * LROW + j] = S[j];
#pragma unroll
  for (int j = 0; j < NDIM; ++j) Vs[t * LROW + j] = (t == j) ? 1.f : 0.f;
  __syncthreads();

  for (int sweep = 0; sweep < NSWEEP; ++sweep) {
    for (int r = 0; r < NDIM - 1; ++r) {
      // --- circle-method pairing: find this index's partner in round r ---
      int partner;
      if (t == NDIM - 1) {
        partner = r;
      } else {
        int k = t - r; k = (k % 63 + 63) % 63;
        if (k == 0)        partner = NDIM - 1;
        else if (k <= 31) { int m = r - k;        if (m < 0)   m += 63; partner = m; }
        else              { int m = r + (63 - k); if (m >= 63) m -= 63; partner = m; }
      }
      const int  p   = t < partner ? t : partner;
      const int  q   = t < partner ? partner : t;
      const bool isP = (t == p);

      // --- rotation (identical on both members of the pair) ---
      const float app = As[p * LROW + p];
      const float aqq = As[q * LROW + q];
      const float apq = As[p * LROW + q];
      float c, s;
      if (fabsf(apq) > 1e-12f) {
        const float tau = (aqq - app) / (2.f * apq);
        float tt = 1.f / (fabsf(tau) + sqrtf(1.f + tau * tau));
        if (tau < 0.f) tt = -tt;
        c = rsqrtf(1.f + tt * tt);
        s = tt * c;
      } else { c = 1.f; s = 0.f; }

      // --- row phase: A <- J^T A (rows disjoint across pairs) ---
      float tr_[NDIM];
      const float* rp = &As[p * LROW];
      const float* rq = &As[q * LROW];
      if (isP) {
#pragma unroll
        for (int j = 0; j < NDIM; ++j) tr_[j] = c * rp[j] - s * rq[j];
      } else {
#pragma unroll
        for (int j = 0; j < NDIM; ++j) tr_[j] = s * rp[j] + c * rq[j];
      }
      __syncthreads();
#pragma unroll
      for (int j = 0; j < NDIM; ++j) As[t * LROW + j] = tr_[j];
      __syncthreads();

      // --- column phase: A <- A J, V <- V J (columns disjoint) ---
      float ca_[NDIM], cv_[NDIM];
      if (isP) {
#pragma unroll
        for (int i = 0; i < NDIM; ++i) {
          ca_[i] = c * As[i * LROW + p] - s * As[i * LROW + q];
          cv_[i] = c * Vs[i * LROW + p] - s * Vs[i * LROW + q];
        }
      } else {
#pragma unroll
        for (int i = 0; i < NDIM; ++i) {
          ca_[i] = s * As[i * LROW + p] + c * As[i * LROW + q];
          cv_[i] = s * Vs[i * LROW + p] + c * Vs[i * LROW + q];
        }
      }
      __syncthreads();
#pragma unroll
      for (int i = 0; i < NDIM; ++i) {
        As[i * LROW + t] = ca_[i];
        Vs[i * LROW + t] = cv_[i];
      }
      __syncthreads();
    }
  }

  // eigenvalues on the diagonal -> log
  lw[t] = logf(fmaxf(As[t * LROW + t], 1e-20f));
  __syncthreads();

  // logX[t][j] = sum_k V[t][k] * log(w_k) * V[j][k]   (row j reads broadcast)
  float wt[NDIM];
#pragma unroll
  for (int k = 0; k < NDIM; ++k) wt[k] = Vs[t * LROW + k] * lw[k];

  if (HALF_OUT) {
    _Float16* O = (_Float16*)dst + mat * NDIM2 + (size_t)t * NDIM;
    for (int j = 0; j < NDIM; ++j) {
      float acc = 0.f;
#pragma unroll
      for (int k = 0; k < NDIM; ++k) acc += wt[k] * Vs[j * LROW + k];
      O[j] = (_Float16)acc;
    }
  } else {
    float* O = (float*)dst + mat * NDIM2 + (size_t)t * NDIM;
    for (int j = 0; j < NDIM; ++j) {
      float acc = 0.f;
#pragma unroll
      for (int k = 0; k < NDIM; ++k) acc += wt[k] * Vs[j * LROW + k];
      O[j] = acc;
    }
  }
}

// ---------------------------------------------------------------------------
// 3) bias[c] = tr(logP_c * A_sym_c) = sum_ij logP[c,ij]*A_sym[c,ij]
// ---------------------------------------------------------------------------
__global__ __launch_bounds__(256) void dot_bias(const float* __restrict__ logP,
                                                const float* __restrict__ As32,
                                                float* __restrict__ bias) {
  __shared__ float red[256];
  const int c = blockIdx.x, t = threadIdx.x;
  const float* lp = logP + (size_t)c * NDIM2;
  const float* as = As32 + (size_t)c * NDIM2;
  float acc = 0.f;
  for (int e = t; e < NDIM2; e += 256) acc += lp[e] * as[e];
  red[t] = acc;
  __syncthreads();
  for (int s2 = 128; s2 > 0; s2 >>= 1) {
    if (t < s2) red[t] += red[t + s2];
    __syncthreads();
  }
  if (t == 0) bias[c] = red[0];
}

// ---------------------------------------------------------------------------
// 4) out[rowBase+m, c] = sum_k logX16[m,k]*Asym16[c,k] - bias[c]
//    Block: 256 threads / 8 waves, M-tile 128 rows, full C=128 per wave.
//    B tile (A_sym, 128x32 f16 per k-chunk) is shared by all 8 waves and
//    staged in LDS with double-buffered GLOBAL_LOAD_ASYNC_TO_LDS_B128
//    (tracked by ASYNCcnt; s_wait_asynccnt + barrier before consumption).
//    Fragment layouts per CDNA5 ISA 7.12.2:
//      A 16x32 f16: lanes 0-15 = rows, VGPR0-3 K=(laneHi?8:0)+0..7,
//                   VGPR4-7 K=16+(laneHi?8:0)+0..7  -> two 16B global loads.
//      B 32x16 f16: lane = column, same K order as A: (laneHi?8:0)+0..7, 16+(laneHi?8:0)+0..7
//                   -> two 16B LDS loads, row stride 80B (bank-spread, aligned).
//      D 16x16 f32: VGPR v, lane L -> row = v + (L>=16 ? 8:0), col = L&15.
// ---------------------------------------------------------------------------
__global__ __launch_bounds__(256) void rmlr_gemm(const _Float16* __restrict__ logX16,
                                                 const _Float16* __restrict__ Asym16,
                                                 const float* __restrict__ bias,
                                                 float* __restrict__ out,
                                                 int rowBase) {
  __shared__ _Float16 Bs[2 * BBUF];

  const int lane   = threadIdx.x & 31;
  const int wave   = threadIdx.x >> 5;
  const int laneLo = lane & 15;
  const int laneHi = lane >> 4;
  const int m0     = blockIdx.x * 128 + wave * 16;   // chunk-local M tile

  const _Float16* arow = logX16 + (size_t)(m0 + laneLo) * NDIM2;

  // Cooperative async B-tile fill: thread t loads 16B of row c = t/2.
  const uint64_t sbase   = (uint64_t)(uintptr_t)Asym16;
  const uint32_t ldsBase = (uint32_t)(uintptr_t)(&Bs[0]);
  const int      bc      = threadIdx.x >> 1;   // 0..127: A_sym row (class)
  const int      bseg    = threadIdx.x & 1;    // 32B segment within 64B row
  auto issueB = [&](int kb, int buf) {
    // global: Asym16[bc][kb*32 + bseg*16 ..] ; lds: Bs[buf][bc*BROW + bseg*16]
    uint32_t goff = (uint32_t)(bc * (NDIM2 * 2) + kb * 64 + bseg * 32);
    uint32_t loff = ldsBase + (uint32_t)(buf * (BBUF * 2) + bc * (BROW * 2) + bseg * 32);
    asm volatile("global_load_async_to_lds_b128 %0, %1, %2"
                 :: "v"(loff), "v"(goff), "s"(sbase) : "memory");
  };

  const v8f vzero = {};
  v8f acc[8];
#pragma unroll
  for (int ct = 0; ct < 8; ++ct) acc[ct] = vzero;

  const int NK = NDIM2 / 32;   // 128 k-chunks
  issueB(0, 0);                // prologue: fill buffer 0

  for (int kb = 0; kb < NK; ++kb) {
    const int buf = kb & 1;
    if (kb + 1 < NK) {
      issueB(kb + 1, buf ^ 1);                          // prefetch next tile
      asm volatile("s_wait_asynccnt 0x2" ::: "memory"); // this wave's buf done
    } else {
      asm volatile("s_wait_asynccnt 0x0" ::: "memory");
    }
    __syncthreads();   // all waves' async fills for `buf` visible in LDS

    const int k0 = kb * 32;
    __builtin_prefetch(arow + k0 + 64, 0, 1);   // global_prefetch_b8 (A stream)

    H16 a;
    a.h[0] = *(const v8h*)(arow + k0 + laneHi * 8);        // VGPR0-3: K 0..7 (+8)
    a.h[1] = *(const v8h*)(arow + k0 + 16 + laneHi * 8);   // VGPR4-7: K16..23 (+8)

    // Preload all 8 B fragments from LDS, then issue WMMAs back-to-back.
    const _Float16* bb = &Bs[0] + buf * BBUF;
    H16 b[8];
#pragma unroll
    for (int ct = 0; ct < 8; ++ct) {
      const int col = ct * 16 + laneLo;
      b[ct].h[0] = *(const v8h*)(bb + col * BROW + laneHi * 8);
      b[ct].h[1] = *(const v8h*)(bb + col * BROW + 16 + laneHi * 8);
    }
#pragma unroll
    for (int ct = 0; ct < 8; ++ct) {
      acc[ct] = __builtin_amdgcn_wmma_f32_16x16x32_f16(
          /*neg_a=*/false, a.v, /*neg_b=*/false, b[ct].v,
          /*c_mod=*/(short)0, acc[ct], /*reuse_a=*/false, /*reuse_b=*/false);
    }
    __syncthreads();   // everyone done reading `buf` before it is refilled
  }

  // Epilogue: D layout -> row = v + laneHi*8, col = laneLo (per 16x16 tile)
#pragma unroll
  for (int ct = 0; ct < 8; ++ct) {
    const int col = ct * 16 + laneLo;
    const float bsub = bias[col];
#pragma unroll
    for (int v = 0; v < 8; ++v) {
      const int row = rowBase + m0 + laneHi * 8 + v;
      out[(size_t)row * NCLS + col] = acc[ct][v] - bsub;
    }
  }
}

// ---------------------------------------------------------------------------
// Host-side orchestration (graph-capture safe: launches only).
// Workspace layout (~37.1 MB):
//   [0,2MB)        As32      128*4096 f32
//   [2MB,3MB)      As16      128*4096 f16
//   [3MB,5MB)      logP32    128*4096 f32
//   [5MB,5MB+512)  bias      128 f32
//   [5MB+4KB, +32MB) logX16  CHUNK*4096 f16 (reused per chunk, L2-resident)
// ---------------------------------------------------------------------------
extern "C" void kernel_launch(void* const* d_in, const int* in_sizes, int n_in,
                              void* d_out, int out_size, void* d_ws, size_t ws_size,
                              hipStream_t stream) {
  (void)in_sizes; (void)n_in; (void)out_size; (void)ws_size;

  const float* X = (const float*)d_in[0];   // (32768, 64, 64) f32
  const float* P = (const float*)d_in[1];   // (128, 64, 64)   f32
  const float* A = (const float*)d_in[2];   // (128, 64, 64)   f32
  float* out = (float*)d_out;               // (32768, 128)    f32

  char* ws = (char*)d_ws;
  float*    As32   = (float*)ws;
  _Float16* As16   = (_Float16*)(ws + (2u << 20));
  float*    logP32 = (float*)(ws + (3u << 20));
  float*    bias   = (float*)(ws + (5u << 20));
  _Float16* logX16 = (_Float16*)(ws + (5u << 20) + 4096);

  prep_asym<<<NCLS, 256, 0, stream>>>(A, As32, As16);
  jacobi_logm<false><<<NCLS, 64, 0, stream>>>(P, (void*)logP32);
  dot_bias<<<NCLS, 256, 0, stream>>>(logP32, As32, bias);

  for (int base = 0; base < NMAT; base += CHUNK) {
    jacobi_logm<true><<<CHUNK, 64, 0, stream>>>(X + (size_t)base * NDIM2,
                                                (void*)logX16);
    rmlr_gemm<<<CHUNK / 128, 256, 0, stream>>>(logX16, As16, bias, out, base);
  }
}
